// FreqSparseAttn_24567212933782
// MI455X (gfx1250) — matrix operand order repaired and checked by reading, not compiled, not run on hardware
//
#include <hip/hip_runtime.h>

// ---------------------------------------------------------------------------
// FreqSparseAttn on gfx1250 (CDNA5, wave32, WMMA 16x16x32 f16 -> f32)
// K-tile staging via Tensor Data Mover (tensor_load_to_lds + s_wait_tensorcnt)
// ---------------------------------------------------------------------------

typedef __attribute__((ext_vector_type(16))) _Float16 v16h;
typedef __attribute__((ext_vector_type(8)))  _Float16 v8h;
typedef __attribute__((ext_vector_type(8)))  float    v8f;
typedef __attribute__((ext_vector_type(4)))  unsigned int v4u;
typedef __attribute__((ext_vector_type(8)))  int      v8i;
typedef __attribute__((ext_vector_type(4)))  int      v4i;

static constexpr int B_  = 2;
static constexpr int L_  = 2048;
static constexpr int D_  = 1024;
static constexpr int H_  = 16;
static constexpr int DH_ = 64;
static constexpr int M_  = B_ * L_;   // 4096 rows

static __device__ __forceinline__ v16h cat8(v8h lo, v8h hi) {
  return __builtin_shufflevector(lo, hi, 0,1,2,3,4,5,6,7,8,9,10,11,12,13,14,15);
}

static __device__ __forceinline__ v8f wmma_f16(v16h a, v16h b, v8f c) {
  // D = A(16x32 f16) * B(32x16 f16) + C(16x16 f32)
  return __builtin_amdgcn_wmma_f32_16x16x32_f16(false, a, false, b,
                                                (short)0, c, false, false);
}

// ---------------------------------------------------------------------------
// Tensor Data Mover: 2D f16 tile (tile_w x tile_h) from global (row stride
// row_stride_elems) into LDS at lds_addr, with LDS row padding
// (pad_interval_code per ISA: 4 -> every 32 DWORDs; pad_amount_code: 3 -> 4
// DWORDs = 16B). D# built per CDNA5 ISA.
// ---------------------------------------------------------------------------
static __device__ __forceinline__ void tdm_load_2d_f16(
    unsigned int lds_addr, const void* gaddr, int tile_w_elems, int tile_h,
    int row_stride_elems, int pad_interval_code, int pad_amount_code) {
  const unsigned long long ga = (unsigned long long)(size_t)gaddr;
  v4u g0;
  g0[0] = 1u;                                   // count=1, user mode, no gather
  g0[1] = lds_addr;                             // LDS byte address
  g0[2] = (unsigned int)ga;                     // global_addr[31:0]
  g0[3] = (unsigned int)((ga >> 32) & 0x01FFFFFFu) | (2u << 30); // [56:32], type=2

  // tensor_dim0 = tensor_dim1 = 2^20 (tiles never out-of-bounds)
  v8i g1;
  g1[0] = (1 << 16)                             // data_size = 2 bytes
        | (1 << 20)                             // pad_enable (LDS row padding)
        | (pad_interval_code << 22)
        | (pad_amount_code << 25);
  g1[1] = 0;                                    // abar=0, tensor_dim0 low16 = 0
  g1[2] = 0x10 | (0 << 16);                     // tensor_dim0 hi16, dim1 lo16
  g1[3] = 0x10 | (tile_w_elems << 16);          // tensor_dim1 hi16, tile_dim0
  g1[4] = tile_h;                               // tile_dim1 (tile_dim2 = 0)
  g1[5] = row_stride_elems;                     // tensor_dim0_stride low32
  g1[6] = 0;
  g1[7] = 0;

  v4i z4 = {0, 0, 0, 0};
#if __has_include(<hip/amd_detail/amd_gfx1250_TDM.h>)
  v8i z8 = {0, 0, 0, 0, 0, 0, 0, 0};
  __builtin_amdgcn_tensor_load_to_lds(g0, g1, z4, z4, z8, 0);
#else
  __builtin_amdgcn_tensor_load_to_lds(g0, g1, z4, z4, 0);
#endif
}

// ---- LDS row loaders: 16 elements -> f16 ----------------------------------
static __device__ __forceinline__ void loadrow(const float* g, _Float16* s) {
#pragma unroll
  for (int j = 0; j < 4; ++j) {
    float4 v = ((const float4*)g)[j];
    s[4*j+0] = (_Float16)v.x; s[4*j+1] = (_Float16)v.y;
    s[4*j+2] = (_Float16)v.z; s[4*j+3] = (_Float16)v.w;
  }
}
static __device__ __forceinline__ void loadrow(const _Float16* g, _Float16* s) {
  *(v8h*)(s)     = *(const v8h*)(g);
  *(v8h*)(s + 8) = *(const v8h*)(g + 8);
}

// ---------------------------------------------------------------------------
// GEMM: Y[M,N] = X[M,K] @ W[N,K]^T + bias[N]
// 256 threads = 8 waves. Block tile 128x128, wave tile 32x64 (2x4 WMMA C).
// ---------------------------------------------------------------------------
template <typename TIN, typename TOUT>
__global__ __launch_bounds__(256) void gemm_xwT_bias(
    const TIN* __restrict__ X, const float* __restrict__ W,
    const float* __restrict__ bias, TOUT* __restrict__ Y,
    int M, int N, int K) {
  __shared__ __align__(16) _Float16 Xs[128][40];   // 40-half row stride (80B)
  __shared__ __align__(16) _Float16 Ws[128][40];

  const int tid  = threadIdx.x;
  const int lane = tid & 31;
  const int wid  = tid >> 5;
  const int half = lane >> 4;
  const int ln   = lane & 15;
  const int wm   = wid & 3;          // 4 wave-rows  * 32
  const int wn   = wid >> 2;         // 2 wave-cols  * 64
  const int lrow = tid >> 1;         // 0..127
  const int lcol = (tid & 1) * 16;

  const int m0 = blockIdx.y * 128;
  const int n0 = blockIdx.x * 128;
  const TIN*   xg = X + (size_t)(m0 + lrow) * K + lcol;
  const float* wg = W + (size_t)(n0 + lrow) * K + lcol;

  v8f acc[2][4] = {};

  for (int kb = 0; kb < K; kb += 32) {
    __syncthreads();
    loadrow(xg + kb, &Xs[lrow][lcol]);
    loadrow(wg + kb, &Ws[lrow][lcol]);
    __syncthreads();

    v16h a[2], bm[4];
#pragma unroll
    for (int mt = 0; mt < 2; ++mt) {   // A frag: row = ln, k = (i&7)+8*half+16*(i>>3)
      int m = wm * 32 + mt * 16 + ln;
      v8h lo = *(const v8h*)&Xs[m][8 * half];
      v8h hi = *(const v8h*)&Xs[m][16 + 8 * half];
      a[mt] = cat8(lo, hi);
    }
#pragma unroll
    for (int nt = 0; nt < 4; ++nt) {   // B frag: col = ln, k = (i&7)+8*half+16*(i>>3)
      int n = wn * 64 + nt * 16 + ln;
      v8h lo = *(const v8h*)&Ws[n][8 * half];
      v8h hi = *(const v8h*)&Ws[n][16 + 8 * half];
      bm[nt] = cat8(lo, hi);
    }
#pragma unroll
    for (int mt = 0; mt < 2; ++mt)
#pragma unroll
      for (int nt = 0; nt < 4; ++nt)
        acc[mt][nt] = wmma_f16(a[mt], bm[nt], acc[mt][nt]);
  }

#pragma unroll
  for (int nt = 0; nt < 4; ++nt) {
    const int gc = n0 + wn * 64 + nt * 16 + ln;
    const float bc = bias[gc];
#pragma unroll
    for (int mt = 0; mt < 2; ++mt) {
#pragma unroll
      for (int r = 0; r < 8; ++r) {    // C/D: row = r + 8*half, col = ln
        const int gr = m0 + wm * 32 + mt * 16 + r + 8 * half;
        Y[(size_t)gr * N + gc] = (TOUT)(acc[mt][nt][r] + bc);
      }
    }
  }
}

// ---------------------------------------------------------------------------
// Energy key mask: bias[b*L+j] = (mean|energy| > 0.1) ? 0 : -inf
// ---------------------------------------------------------------------------
__global__ __launch_bounds__(256) void energy_mask(
    const float* __restrict__ E, float* __restrict__ mb, int rows, int D) {
  const int row  = blockIdx.x * 8 + (threadIdx.x >> 5);
  const int lane = threadIdx.x & 31;
  if (row >= rows) return;
  const float* e = E + (size_t)row * D;
  float s = 0.f;
  for (int i = lane; i < D; i += 32) s += fabsf(e[i]);
#pragma unroll
  for (int d = 16; d >= 1; d >>= 1) s += __shfl_xor(s, d, 32);
  if (lane == 0)
    mb[row] = (s * (1.0f / (float)D) > 0.1f) ? 0.0f : -__builtin_inff();
}

// ---------------------------------------------------------------------------
// Flash-style masked attention per (b,h).
// 8 waves x 16 queries per block; key blocks of 32 shared in LDS.
// K tile staged by the Tensor Data Mover (padded-row 2D DMA); V staged
// manually with an on-the-fly transpose (TDM cannot transpose).
// ---------------------------------------------------------------------------
__global__ __launch_bounds__(256) void attn_fsparse(
    const _Float16* __restrict__ Qh, const _Float16* __restrict__ Kh,
    const _Float16* __restrict__ Vh, const float* __restrict__ mbias,
    _Float16* __restrict__ Oh) {
  const float NEG_INF = -__builtin_inff();

  __shared__ __align__(16) _Float16 Ks[32][72];      // [key][dh], 144B rows (TDM pad)
  __shared__ __align__(16) _Float16 VT[64][40];      // [dh][key]  (transposed V)
  __shared__ __align__(16) _Float16 Ps[8][16][40];   // per-wave prob scratch
  __shared__ float Ms[32];                           // per-key mask bias

  const int tid  = threadIdx.x;
  const int lane = tid & 31;
  const int wid  = tid >> 5;
  const int half = lane >> 4;
  const int ln   = lane & 15;
  const int bh = blockIdx.y, b = bh >> 4, h = bh & 15;
  const int q0 = blockIdx.x * 128 + wid * 16;

  // Q A-fragments (dh=64 -> two K=32 chunks), loaded once from global f16
  const _Float16* qp = Qh + ((size_t)(b * L_ + q0 + ln)) * D_ + h * DH_;
  v16h aq[2];
#pragma unroll
  for (int c = 0; c < 2; ++c) {
    v8h lo = *(const v8h*)(qp + 32 * c + 8 * half);
    v8h hi = *(const v8h*)(qp + 32 * c + 16 + 8 * half);
    aq[c] = cat8(lo, hi);
  }

  v8f Oc[4] = {};
  float mrow[8], lrow[8];
#pragma unroll
  for (int r = 0; r < 8; ++r) { mrow[r] = NEG_INF; lrow[r] = 0.f; }

  const int key_t = tid >> 3;        // 0..31
  const int off_t = (tid & 7) * 8;   // 0..56
  const size_t kv0 = (size_t)b * L_ * D_ + (size_t)h * DH_;
  const unsigned int ks_lds = (unsigned int)(size_t)&Ks[0][0];

  for (int j0 = 0; j0 < L_; j0 += 32) {
    __syncthreads();
    // --- stage V^T [64 x 32] manually; K [32 x 64] via TDM ---
    v8h vv = *(const v8h*)(Vh + kv0 + (size_t)(j0 + key_t) * D_ + off_t);
#pragma unroll
    for (int j = 0; j < 8; ++j) VT[off_t + j][key_t] = vv[j];
    if (tid < 32) Ms[tid] = mbias[b * L_ + j0 + tid];
    if (j0 + 32 < L_) {   // L2 prefetch of the next key block
      __builtin_prefetch(Kh + kv0 + (size_t)(j0 + 32 + key_t) * D_ + off_t, 0, 3);
      __builtin_prefetch(Vh + kv0 + (size_t)(j0 + 32 + key_t) * D_ + off_t, 0, 3);
    }
    if (wid == 0) {
      // 32 rows x 64 halfs, memory row stride D_; LDS rows 128B + 16B pad
      tdm_load_2d_f16(ks_lds, Kh + kv0 + (size_t)j0 * D_,
                      /*tile_w=*/64, /*tile_h=*/32, /*stride=*/D_,
                      /*pad_interval(32 DW)=*/4, /*pad_amount(4 DW)=*/3);
      __builtin_amdgcn_s_wait_tensorcnt(0);
    }
    __syncthreads();

    // --- scores S = Q K^T for two 16-key tiles (K-dim = dh = 2 x 32) ---
    v8f S[2];
#pragma unroll
    for (int jt = 0; jt < 2; ++jt) {
      v8f s = {};
#pragma unroll
      for (int c = 0; c < 2; ++c) {
        v8h lo = *(const v8h*)&Ks[jt * 16 + ln][32 * c + 8 * half];
        v8h hi = *(const v8h*)&Ks[jt * 16 + ln][32 * c + 16 + 8 * half];
        s = wmma_f16(aq[c], cat8(lo, hi), s);
      }
      S[jt] = s;
    }

    // --- online softmax with -inf key mask ---
    const float mb0 = Ms[ln], mb1 = Ms[16 + ln];
#pragma unroll
    for (int r = 0; r < 8; ++r) {
      float s0 = S[0][r] * 0.125f + mb0;   // 1/sqrt(64)
      float s1 = S[1][r] * 0.125f + mb1;
      float rmax = fmaxf(s0, s1);
#pragma unroll
      for (int d = 8; d >= 1; d >>= 1) rmax = fmaxf(rmax, __shfl_xor(rmax, d, 32));
      const float mold = mrow[r];
      const float mnew = fmaxf(mold, rmax);
      float alpha, p0, p1;
      if (mnew == NEG_INF) {               // every key so far masked
        alpha = 1.f; p0 = 0.f; p1 = 0.f;
      } else {
        alpha = __expf(mold - mnew);
        p0 = __expf(s0 - mnew);
        p1 = __expf(s1 - mnew);
      }
      float rsum = p0 + p1;
#pragma unroll
      for (int d = 8; d >= 1; d >>= 1) rsum += __shfl_xor(rsum, d, 32);
      lrow[r] = lrow[r] * alpha + rsum;
      mrow[r] = mnew;
#pragma unroll
      for (int t = 0; t < 4; ++t) Oc[t][r] *= alpha;
      const int m = r + 8 * half;
      Ps[wid][m][ln]      = (_Float16)p0;  // C-layout -> LDS
      Ps[wid][m][16 + ln] = (_Float16)p1;
    }

    // --- probs back as A fragment; O += P(16x32) * V(32x64) ---
    v8h plo = *(const v8h*)&Ps[wid][ln][8 * half];
    v8h phi = *(const v8h*)&Ps[wid][ln][16 + 8 * half];
    v16h pa = cat8(plo, phi);
#pragma unroll
    for (int t = 0; t < 4; ++t) {
      v8h lo = *(const v8h*)&VT[t * 16 + ln][8 * half];
      v8h hi = *(const v8h*)&VT[t * 16 + ln][16 + 8 * half];
      Oc[t] = wmma_f16(pa, cat8(lo, hi), Oc[t]);
    }
  }

  // epilogue: O / l  (fully-masked rows -> 0, matches nan_to_num)
#pragma unroll
  for (int t = 0; t < 4; ++t)
#pragma unroll
    for (int r = 0; r < 8; ++r) {
      const float l = lrow[r];
      const float o = (l > 0.f) ? (Oc[t][r] / l) : 0.f;
      const int q = q0 + r + 8 * half;
      Oh[((size_t)(b * L_ + q)) * D_ + h * DH_ + t * 16 + ln] = (_Float16)o;
    }
}

// ---------------------------------------------------------------------------
extern "C" void kernel_launch(void* const* d_in, const int* in_sizes, int n_in,
                              void* d_out, int out_size, void* d_ws, size_t ws_size,
                              hipStream_t stream) {
  const float* qx = (const float*)d_in[0];
  const float* kx = (const float*)d_in[1];
  const float* vx = (const float*)d_in[2];
  const float* en = (const float*)d_in[3];
  const float* Wq = (const float*)d_in[4];
  const float* bq = (const float*)d_in[5];
  const float* Wk = (const float*)d_in[6];
  const float* bk = (const float*)d_in[7];
  const float* Wv = (const float*)d_in[8];
  const float* bv = (const float*)d_in[9];
  const float* Wo = (const float*)d_in[10];
  const float* bo = (const float*)d_in[11];
  float* out = (float*)d_out;

  // workspace: Qh | Kh | Vh | Ah (f16, 8 MB each) | mask bias (f32)
  _Float16* Qh = (_Float16*)d_ws;
  _Float16* Kh = Qh + (size_t)M_ * D_;
  _Float16* Vh = Kh + (size_t)M_ * D_;
  _Float16* Ah = Vh + (size_t)M_ * D_;
  float*    mb = (float*)(Ah + (size_t)M_ * D_);

  const dim3 gg(D_ / 128, M_ / 128);   // (8, 32)
  gemm_xwT_bias<float, _Float16><<<gg, 256, 0, stream>>>(qx, Wq, bq, Qh, M_, D_, D_);
  gemm_xwT_bias<float, _Float16><<<gg, 256, 0, stream>>>(kx, Wk, bk, Kh, M_, D_, D_);
  gemm_xwT_bias<float, _Float16><<<gg, 256, 0, stream>>>(vx, Wv, bv, Vh, M_, D_, D_);

  energy_mask<<<M_ / 8, 256, 0, stream>>>(en, mb, M_, D_);

  attn_fsparse<<<dim3(L_ / 128, B_ * H_), 256, 0, stream>>>(Qh, Kh, Vh, mb, Ah);

  gemm_xwT_bias<_Float16, float><<<gg, 256, 0, stream>>>(Ah, Wo, bo, out, M_, D_, D_);
}
